// CrossAttention_7292854468702
// MI455X (gfx1250) — matrix operand order repaired and checked by reading, not compiled, not run on hardware
//
#include <hip/hip_runtime.h>

#define B_    8
#define H_    12
#define LQ    1024
#define LKV   1024
#define DIM_  768
#define DH    64
#define SCALE 0.125f   // 64^-0.5

typedef __attribute__((ext_vector_type(16))) __bf16 v16bf;
typedef __attribute__((ext_vector_type(8)))  __bf16 v8bf;
typedef __attribute__((ext_vector_type(8)))  float  v8f;
typedef __attribute__((ext_vector_type(4)))  float  v4f;

static __device__ __forceinline__ __bf16 f2bf(float x) {
  return (__bf16)x;   // backend: v_cvt_pk_bf16_f32 / bf16 cvt, RTNE
}

static __device__ __forceinline__ v8bf ld8(const __bf16* p) {
  return *(const v8bf*)p;
}

static __device__ __forceinline__ v16bf mk16(v8bf lo, v8bf hi) {
  v16bf r;
#pragma unroll
  for (int i = 0; i < 8; ++i) { r[i] = lo[i]; r[8 + i] = hi[i]; }
  return r;
}

static __device__ __forceinline__ v8f wmma_bf16(v16bf a, v16bf b, v8f c) {
  return __builtin_amdgcn_wmma_f32_16x16x32_bf16(false, a, false, b,
                                                 (short)0, c, false, false);
}

// ---------------------------------------------------------------------------
// Kernel 0: one-pass fp32 -> bf16 conversion (8 elems / thread, b128 traffic)
// ---------------------------------------------------------------------------
__global__ void cvt_bf16_kernel(const float* __restrict__ in,
                                __bf16* __restrict__ outp) {
  const size_t i = ((size_t)blockIdx.x * blockDim.x + threadIdx.x) * 8;
  v4f a = *(const v4f*)(in + i);
  v4f b = *(const v4f*)(in + i + 4);
  v8bf r;
#pragma unroll
  for (int j = 0; j < 4; ++j) { r[j] = f2bf(a[j]); r[4 + j] = f2bf(b[j]); }
  *(v8bf*)(outp + i) = r;
}

// ---------------------------------------------------------------------------
// Scatter helper for the fused QKV projection outputs.
// ---------------------------------------------------------------------------
static __device__ __forceinline__ void store_proj(bool isQ, int ncol, int m,
                                                  float v,
                                                  __bf16* __restrict__ qh,
                                                  __bf16* __restrict__ kb,
                                                  __bf16* __restrict__ vtb) {
  const int bb = m >> 10;        // / 1024
  const int rr = m & 1023;
  if (isQ) {
    const int hh = ncol >> 6, dd = ncol & 63;
    qh[(((size_t)bb * H_ + hh) * LQ + rr) * DH + dd] = f2bf(v);
  } else if (ncol < DIM_) {
    const int hh = ncol >> 6, dd = ncol & 63;
    kb[(((size_t)bb * H_ + hh) * LKV + rr) * DH + dd] = f2bf(v);
  } else {
    const int nc = ncol - DIM_;
    const int hh = nc >> 6, dd = nc & 63;
    vtb[(((size_t)bb * H_ + hh) * DH + dd) * LKV + rr] = f2bf(v);
  }
}

// ---------------------------------------------------------------------------
// Kernel 1: fused Q / KV projections from pre-converted bf16 inputs.
// Wave = 32x32 output tile (4 WMMA / k-step, A/B frags reused twice).
// grid = (256 M-tiles, 18), block = 128 (4 waves, one N-tile each).
// ntile < 24  -> q @ Wq^T   -> qh (B,H,Lq,64)
// ntile >= 24 -> kv @ Wkv^T -> K (B,H,Lkv,64) or V^T (B,H,64,Lkv)
// ---------------------------------------------------------------------------
__global__ void proj_qkv_kernel(const __bf16* __restrict__ qbf,
                                const __bf16* __restrict__ kvbf,
                                const __bf16* __restrict__ wqbf,
                                const __bf16* __restrict__ wkvbf,
                                __bf16* __restrict__ qh,
                                __bf16* __restrict__ kb,
                                __bf16* __restrict__ vtb) {
  const int wid  = threadIdx.x >> 5;
  const int lane = threadIdx.x & 31;
  const int nl   = lane & 15;
  const int half = lane >> 4;
  const int koff = half * 8;

  const int mtile = blockIdx.x;
  const int ntile = blockIdx.y * 4 + wid;
  const bool isQ  = ntile < 24;
  const int  nt   = isQ ? ntile : ntile - 24;

  const __bf16* A = isQ ? qbf : kvbf;
  const __bf16* W = isQ ? wqbf : wkvbf;
  const __bf16* a0 = A + (size_t)(mtile * 32 + nl) * DIM_;
  const __bf16* a1 = a0 + 16 * DIM_;
  const __bf16* w0 = W + (size_t)(nt * 32 + nl) * DIM_;
  const __bf16* w1 = w0 + 16 * DIM_;

  v8f c00 = {}, c01 = {}, c10 = {}, c11 = {};
  for (int k0 = 0; k0 < DIM_; k0 += 32) {
    v16bf A0 = mk16(ld8(a0 + k0 + koff), ld8(a0 + k0 + 16 + koff));
    v16bf A1 = mk16(ld8(a1 + k0 + koff), ld8(a1 + k0 + 16 + koff));
    v16bf B0 = mk16(ld8(w0 + k0 + koff), ld8(w0 + k0 + 16 + koff));
    v16bf B1 = mk16(ld8(w1 + k0 + koff), ld8(w1 + k0 + 16 + koff));
    c00 = wmma_bf16(A0, B0, c00);
    c01 = wmma_bf16(A0, B1, c01);
    c10 = wmma_bf16(A1, B0, c10);
    c11 = wmma_bf16(A1, B1, c11);
  }

  const int ncol0 = nt * 32 + nl;
#pragma unroll
  for (int r = 0; r < 8; ++r) {
    const int m0 = mtile * 32 + half * 8 + r;
    store_proj(isQ, ncol0,      m0,      c00[r], qh, kb, vtb);
    store_proj(isQ, ncol0 + 16, m0,      c01[r], qh, kb, vtb);
    store_proj(isQ, ncol0,      m0 + 16, c10[r], qh, kb, vtb);
    store_proj(isQ, ncol0 + 16, m0 + 16, c11[r], qh, kb, vtb);
  }
}

// ---------------------------------------------------------------------------
// Kernel 2: flash attention. grid = (Lq/128 = 8, B*H = 96), block = 256.
// Each of 8 waves owns 16 query rows; WG stages 32 KV rows per step in LDS.
// ---------------------------------------------------------------------------
__global__ void attn_kernel(const __bf16* __restrict__ qh,
                            const __bf16* __restrict__ kb,
                            const __bf16* __restrict__ vtb,
                            const float* __restrict__ pos,
                            __bf16* __restrict__ xout) {
  __shared__ __bf16 Kt[32][DH];       // [kv row][d]
  __shared__ __bf16 Vt[DH][32];       // [d][kv]
  __shared__ __bf16 Pt[8][16][32];    // per-wave P staging

  const int tid  = threadIdx.x;
  const int wid  = tid >> 5;
  const int lane = tid & 31;
  const int nl   = lane & 15;
  const int half = lane >> 4;
  const int koff = half * 8;

  const int bh = blockIdx.y;
  const int b  = bh / H_;
  const int h  = bh % H_;
  const int qr0 = blockIdx.x * 128 + wid * 16;

  // Preload Q A-fragments for k0 = 0 and 32 (invariant over the KV loop)
  const __bf16* qrow = qh + (((size_t)bh) * LQ + (qr0 + nl)) * DH;
  const v16bf aq0 = mk16(ld8(qrow + koff), ld8(qrow + 16 + koff));
  const v16bf aq1 = mk16(ld8(qrow + 32 + koff), ld8(qrow + 48 + koff));

  float mrun[8], lrun[8];
  v8f xa[4];
#pragma unroll
  for (int r = 0; r < 8; ++r) { mrun[r] = -3.0e38f; lrun[r] = 0.0f; }
#pragma unroll
  for (int t = 0; t < 4; ++t) xa[t] = (v8f){};

  const __bf16* kbase = kb  + (size_t)bh * LKV * DH;
  const __bf16* vbase = vtb + (size_t)bh * DH * LKV;
  const float*  pbase = pos + (size_t)h * LQ * LKV;   // attn_pos batch dim == 1

  for (int j = 0; j < LKV; j += 32) {
    __syncthreads();
    {   // cooperative stage of K tile (32x64) and V^T tile (64x32)
      const int kvr = tid >> 3, dc = (tid & 7) * 8;
      *(v8bf*)(&Kt[kvr][dc]) = ld8(kbase + (size_t)(j + kvr) * DH + dc);
      const int dr = tid >> 2, kc = (tid & 3) * 8;
      *(v8bf*)(&Vt[dr][kc]) = ld8(vbase + (size_t)dr * LKV + j + kc);
    }
    __syncthreads();

    // S = Q K^T : two 16x16 column tiles (j..j+15, j+16..j+31)
    v8f c0 = {}, c1 = {};
    {
      const __bf16* kr0 = &Kt[nl][koff];
      const __bf16* kr1 = &Kt[nl + 16][koff];
      v16bf bA = mk16(ld8(kr0), ld8(kr0 + 16));
      c0 = wmma_bf16(aq0, bA, c0);
      bA = mk16(ld8(kr0 + 32), ld8(kr0 + 48));
      c0 = wmma_bf16(aq1, bA, c0);
      v16bf bB = mk16(ld8(kr1), ld8(kr1 + 16));
      c1 = wmma_bf16(aq0, bB, c1);
      bB = mk16(ld8(kr1 + 32), ld8(kr1 + 48));
      c1 = wmma_bf16(aq1, bB, c1);
    }

    // online softmax over this 16x32 logit block
#pragma unroll
    for (int r = 0; r < 8; ++r) {
      const int row = qr0 + half * 8 + r;
      const float* prow = pbase + (size_t)row * LKV + j;
      float s0 = c0[r] * SCALE + prow[nl];
      float s1 = c1[r] * SCALE + prow[16 + nl];
      float mx = fmaxf(s0, s1);
#pragma unroll
      for (int o = 1; o < 16; o <<= 1) mx = fmaxf(mx, __shfl_xor(mx, o, 32));
      const float mn = fmaxf(mrun[r], mx);
      const float alpha = __expf(mrun[r] - mn);
      const float p0 = __expf(s0 - mn);
      const float p1 = __expf(s1 - mn);
      float sum = p0 + p1;
#pragma unroll
      for (int o = 1; o < 16; o <<= 1) sum += __shfl_xor(sum, o, 32);
      lrun[r] = lrun[r] * alpha + sum;
      mrun[r] = mn;
#pragma unroll
      for (int t = 0; t < 4; ++t) xa[t][r] *= alpha;
      const int R = half * 8 + r;
      Pt[wid][R][nl]      = f2bf(p0);
      Pt[wid][R][16 + nl] = f2bf(p1);
    }
    asm volatile("s_wait_dscnt 0" ::: "memory");   // wave-local LDS RAW

    // x += P @ V  (K = 32, four 16-wide d tiles)
    const v16bf ap = mk16(ld8(&Pt[wid][nl][koff]), ld8(&Pt[wid][nl][16 + koff]));
#pragma unroll
    for (int t = 0; t < 4; ++t) {
      const __bf16* vp = &Vt[t * 16 + nl][koff];
      v16bf bv = mk16(ld8(vp), ld8(vp + 16));
      xa[t] = wmma_bf16(ap, bv, xa[t]);
    }
  }

  // finalize: divide by l, store bf16 x in (B, Lq, H*64) layout
#pragma unroll
  for (int r = 0; r < 8; ++r) {
    const float inv = 1.0f / lrun[r];
    const int qr = qr0 + half * 8 + r;
    const size_t base = ((size_t)b * LQ + qr) * DIM_ + h * DH;
#pragma unroll
    for (int t = 0; t < 4; ++t)
      xout[base + t * 16 + nl] = f2bf(xa[t][r] * inv);
  }
}

// ---------------------------------------------------------------------------
// Kernel 3: out = x @ Wproj^T + bproj (fp32 out). Wave = 32x32 tile.
// grid = (256, 6), block = 128.
// ---------------------------------------------------------------------------
__global__ void outproj_kernel(const __bf16* __restrict__ xb,
                               const __bf16* __restrict__ wpbf,
                               const float* __restrict__ bias,
                               float* __restrict__ out) {
  const int wid  = threadIdx.x >> 5;
  const int lane = threadIdx.x & 31;
  const int nl   = lane & 15;
  const int half = lane >> 4;
  const int koff = half * 8;

  const int mtile = blockIdx.x;
  const int ntile = blockIdx.y * 4 + wid;

  const __bf16* a0 = xb + (size_t)(mtile * 32 + nl) * DIM_;
  const __bf16* a1 = a0 + 16 * DIM_;
  const __bf16* w0 = wpbf + (size_t)(ntile * 32 + nl) * DIM_;
  const __bf16* w1 = w0 + 16 * DIM_;

  v8f c00 = {}, c01 = {}, c10 = {}, c11 = {};
  for (int k0 = 0; k0 < DIM_; k0 += 32) {
    v16bf A0 = mk16(ld8(a0 + k0 + koff), ld8(a0 + k0 + 16 + koff));
    v16bf A1 = mk16(ld8(a1 + k0 + koff), ld8(a1 + k0 + 16 + koff));
    v16bf B0 = mk16(ld8(w0 + k0 + koff), ld8(w0 + k0 + 16 + koff));
    v16bf B1 = mk16(ld8(w1 + k0 + koff), ld8(w1 + k0 + 16 + koff));
    c00 = wmma_bf16(A0, B0, c00);
    c01 = wmma_bf16(A0, B1, c01);
    c10 = wmma_bf16(A1, B0, c10);
    c11 = wmma_bf16(A1, B1, c11);
  }

  const int ncol0 = ntile * 32 + nl;
  const float bs0 = bias[ncol0];
  const float bs1 = bias[ncol0 + 16];
#pragma unroll
  for (int r = 0; r < 8; ++r) {
    const int m0 = mtile * 32 + half * 8 + r;
    out[(size_t)m0 * DIM_ + ncol0]             = c00[r] + bs0;
    out[(size_t)m0 * DIM_ + ncol0 + 16]        = c01[r] + bs1;
    out[(size_t)(m0 + 16) * DIM_ + ncol0]      = c10[r] + bs0;
    out[(size_t)(m0 + 16) * DIM_ + ncol0 + 16] = c11[r] + bs1;
  }
}

// ---------------------------------------------------------------------------
extern "C" void kernel_launch(void* const* d_in, const int* in_sizes, int n_in,
                              void* d_out, int out_size, void* d_ws, size_t ws_size,
                              hipStream_t stream) {
  const float* q    = (const float*)d_in[0];
  const float* kv   = (const float*)d_in[1];
  const float* pos  = (const float*)d_in[2];
  const float* Wq   = (const float*)d_in[3];
  const float* Wkv  = (const float*)d_in[4];
  const float* Wp   = (const float*)d_in[5];
  const float* bp   = (const float*)d_in[6];
  float* out = (float*)d_out;

  const size_t N1 = (size_t)B_ * H_ * LQ * DH;   // 6,291,456
  const size_t NWQ = (size_t)DIM_ * DIM_;        //   589,824
  const size_t NWKV = 2 * NWQ;                   // 1,179,648

  __bf16* qh   = (__bf16*)d_ws;        // bf16 Q heads   (B,H,Lq,64)
  __bf16* kb   = qh + N1;              // bf16 K heads   (B,H,Lkv,64)
  __bf16* vtb  = kb + N1;              // bf16 V^T heads (B,H,64,Lkv)
  __bf16* xb   = vtb + N1;             // bf16 attention output (B,Lq,768)
  __bf16* qbf  = xb + N1;              // bf16 copy of q
  __bf16* kvbf = qbf + N1;             // bf16 copy of kv
  __bf16* wqbf = kvbf + N1;            // bf16 Wq
  __bf16* wkvbf = wqbf + NWQ;          // bf16 Wkv
  __bf16* wpbf = wkvbf + NWKV;         // bf16 Wproj     (total ~80 MB)

  // 0) one-pass fp32 -> bf16 conversions (2048 elems / block)
  cvt_bf16_kernel<<<dim3(N1 / 2048), dim3(256), 0, stream>>>(q, qbf);
  cvt_bf16_kernel<<<dim3(N1 / 2048), dim3(256), 0, stream>>>(kv, kvbf);
  cvt_bf16_kernel<<<dim3(NWQ / 2048), dim3(256), 0, stream>>>(Wq, wqbf);
  cvt_bf16_kernel<<<dim3(NWKV / 2048), dim3(256), 0, stream>>>(Wkv, wkvbf);
  cvt_bf16_kernel<<<dim3(NWQ / 2048), dim3(256), 0, stream>>>(Wp, wpbf);

  // 1) QKV projections: 256 M-tiles (32 rows) x (24 Q + 48 KV) N-tiles
  proj_qkv_kernel<<<dim3(256, 18), dim3(128), 0, stream>>>(qbf, kvbf, wqbf,
                                                           wkvbf, qh, kb, vtb);
  // 2) attention: 8 query blocks of 128 rows x 96 (b,h) pairs
  attn_kernel<<<dim3(LQ / 128, B_ * H_), dim3(256), 0, stream>>>(qh, kb, vtb,
                                                                 pos, xb);
  // 3) output projection: 256 M-tiles x 24 N-tiles
  outproj_kernel<<<dim3(256, 6), dim3(128), 0, stream>>>(xb, wpbf, bp, out);
}
